// ChebNet_BenQin_64364379898363
// MI455X (gfx1250) — matrix operand order repaired and checked by reading, not compiled, not run on hardware
//
#include <hip/hip_runtime.h>
#include <math.h>

// ---------------------------------------------------------------------------
// ChebNet (magnetic Laplacian, K=2) forward for MI455X / gfx1250.
// Big complex GEMMs: v_wmma_f32_16x16x32_f16, f32 accumulate, LDS-staged
// double-buffered tiles (block tile 64x128, wave tile 16x64).
// Lc stored f16 (64MB -> resident in the 192MB L2).
// ---------------------------------------------------------------------------

#define NN   4096      // nodes
#define INC  256       // in channels (== F)
#define FF   256       // filters
#define CC   40        // classes
#define TWO_PI_F 6.28318530717958647692f

typedef _Float16 v8h  __attribute__((ext_vector_type(8)));
typedef _Float16 v16h __attribute__((ext_vector_type(16)));
typedef float    v8f  __attribute__((ext_vector_type(8)));

// Build a 16-half WMMA fragment from two 16-byte chunks (global or LDS).
__device__ __forceinline__ v16h frag2(const _Float16* lo, const _Float16* hi) {
    v8h a = *(const v8h*)lo;
    v8h b = *(const v8h*)hi;
    v16h r;
#pragma unroll
    for (int t = 0; t < 8; ++t) { r[t] = a[t]; r[t + 8] = b[t]; }
    return r;
}

#define WMMA_F16(A, B, Cacc) \
    __builtin_amdgcn_wmma_f32_16x16x32_f16(false, (A), false, (B), (short)0, (Cacc), false, false)

// ---------------------------------------------------------------------------
// Graph construction
// ---------------------------------------------------------------------------

__global__ void k_scatter(const int* __restrict__ edges,
                          const float* __restrict__ w,
                          float* __restrict__ A, int E) {
    int e = blockIdx.x * blockDim.x + threadIdx.x;
    if (e >= E) return;
    int r = edges[e];
    int c = edges[E + e];
    atomicAdd(&A[(size_t)r * NN + c], w[e]);
}

__global__ void k_rowsum(const float* __restrict__ A, float* __restrict__ rs) {
    __shared__ float sm[256];
    int row = blockIdx.x;
    const float* ap = A + (size_t)row * NN;
    float s = 0.f;
    for (int j = threadIdx.x; j < NN; j += 256) s += ap[j];
    sm[threadIdx.x] = s;
    __syncthreads();
    for (int off = 128; off > 0; off >>= 1) {
        if (threadIdx.x < off) sm[threadIdx.x] += sm[threadIdx.x + off];
        __syncthreads();
    }
    if (threadIdx.x == 0) rs[row] = sm[0];
}

__global__ void k_dinv(const float* __restrict__ A,
                       const float* __restrict__ rs,
                       float* __restrict__ dinv) {
    int j = blockIdx.x * 256 + threadIdx.x;           // coalesced column sweep
    float s = 0.f;
    for (int i = 0; i < NN; ++i) s += A[(size_t)i * NN + j];
    float d = 0.5f * (s + rs[j]);                     // d_j = sum_i A_sym[i][j]
    dinv[j] = (d == 0.f) ? 1.f : 1.f / sqrtf(d);
}

// conj(L) = -A_n * exp(-i*Theta): real = -an*cos(th), imag = +an*sin(th)
__global__ void k_build_lc(const float* __restrict__ A,
                           const float* __restrict__ dinv,
                           const float* __restrict__ qp,
                           _Float16* __restrict__ LcR,
                           _Float16* __restrict__ LcI) {
    size_t idx = (size_t)blockIdx.x * 256 + threadIdx.x;  // 4096*4096 threads
    int i = (int)(idx >> 12);
    int j = (int)(idx & 4095);
    float a  = A[idx];
    float at = A[(size_t)j * NN + i];
    float an = dinv[i] * 0.5f * (a + at) * dinv[j];
    float th = TWO_PI_F * (*qp) * (a - at);
    float s, c;
    __sincosf(th, &s, &c);
    LcR[idx] = (_Float16)(-an * c);
    LcI[idx] = (_Float16)( an * s);
}

// ---------------------------------------------------------------------------
// Activation conversion: f32 [N,F] -> f16 row-major + f16 column-major
// ---------------------------------------------------------------------------

__global__ void k_cvt(const float* __restrict__ Xr, const float* __restrict__ Xi,
                      _Float16* __restrict__ rmR, _Float16* __restrict__ rmI,
                      _Float16* __restrict__ cmR, _Float16* __restrict__ cmI) {
    int idx = blockIdx.x * 256 + threadIdx.x;   // N*F
    int r = idx >> 8, c = idx & 255;
    float xr = Xr[idx], xi = Xi[idx];
    rmR[idx] = (_Float16)xr;
    rmI[idx] = (_Float16)xi;
    cmR[(size_t)c * NN + r] = (_Float16)xr;
    cmI[(size_t)c * NN + r] = (_Float16)xi;
}

// ---------------------------------------------------------------------------
// Big complex GEMM: Y[N,F] = Lc[N,N] @ X[N,F].
// Lc f16 row-major, X f16 column-major.
// Block: 256 threads = 8 waves; block tile 64 rows x 128 cols.
// Wave (wy,wx): row tile wy (of 4), col tiles wx*4..wx*4+3 (of 8).
// K-slices of both operands staged in double-buffered LDS (pad 40 halves
// per row -> conflict-free ds_load_b128 fragment reads).
// ---------------------------------------------------------------------------

#define PAD 40   // halves per LDS row (32 data + 8 pad)

__global__ __launch_bounds__(256) void k_cgemm_lc(
        const _Float16* __restrict__ LcR, const _Float16* __restrict__ LcI,
        const _Float16* __restrict__ BcmR, const _Float16* __restrict__ BcmI,
        float* __restrict__ Yr, float* __restrict__ Yi) {
    __shared__ alignas(16) _Float16 sA[2][2][64][PAD];    // [buf][mat][row][k]  20 KB
    __shared__ alignas(16) _Float16 sB[2][2][128][PAD];   // [buf][mat][col][k]  40 KB

    const int tid  = threadIdx.x;
    const int lane = tid & 31;
    const int wave = tid >> 5;
    const int rowBase = (blockIdx.x >> 1) * 64;   // 64 row groups
    const int colBase = (blockIdx.x & 1) * 128;   // 2 col groups
    const int wy = wave >> 1;                     // 0..3 row tile in block
    const int wx = wave & 1;                      // 0..1 col half in block
    const int m  = lane & 15;
    const int hi = lane >> 4;
    const int aoff = hi * 8;                      // A-frag K sub-offset (halves)
    const int boff = hi * 8;                      // B-frag K sub-offset (halves)

    v8f rr[4], ii[4], ri[4], ir[4];
#pragma unroll
    for (int t = 0; t < 4; ++t) { rr[t] = (v8f){}; ii[t] = (v8f){}; ri[t] = (v8f){}; ir[t] = (v8f){}; }

    // Cooperative stage of one K=32 slice of A (64 rows) and B (128 cols).
    auto stage = [&](int p, int k0) {
#pragma unroll
        for (int s = 0; s < 2; ++s) {             // A: 512 16B chunks / 256 thr
            int t   = tid + s * 256;
            int mat = t >> 8;
            int rem = t & 255;
            int row = rem >> 2, ch = rem & 3;
            const _Float16* src = (mat ? LcI : LcR) + (size_t)(rowBase + row) * NN + k0 + ch * 8;
            *(v8h*)&sA[p][mat][row][ch * 8] = *(const v8h*)src;
        }
#pragma unroll
        for (int s = 0; s < 4; ++s) {             // B: 1024 16B chunks / 256 thr
            int t   = tid + s * 256;
            int mat = t >> 9;
            int rem = t & 511;
            int col = rem >> 2, ch = rem & 3;
            const _Float16* src = (mat ? BcmI : BcmR) + (size_t)(colBase + col) * NN + k0 + ch * 8;
            *(v8h*)&sB[p][mat][col][ch * 8] = *(const v8h*)src;
        }
    };

    stage(0, 0);
    __syncthreads();
    for (int k0 = 0; k0 < NN; k0 += 32) {
        int p = (k0 >> 5) & 1;
        if (k0 + 32 < NN) stage(p ^ 1, k0 + 32);  // prefetch next slice into other buffer

        int arow = wy * 16 + m;
        v16h Ar = frag2(&sA[p][0][arow][aoff], &sA[p][0][arow][aoff + 16]);
        v16h Ai = frag2(&sA[p][1][arow][aoff], &sA[p][1][arow][aoff + 16]);
#pragma unroll
        for (int ct = 0; ct < 4; ++ct) {
            int col = wx * 64 + ct * 16 + m;
            v16h Br = frag2(&sB[p][0][col][boff], &sB[p][0][col][boff + 16]);
            v16h Bi = frag2(&sB[p][1][col][boff], &sB[p][1][col][boff + 16]);
            rr[ct] = WMMA_F16(Ar, Br, rr[ct]);
            ii[ct] = WMMA_F16(Ai, Bi, ii[ct]);
            ri[ct] = WMMA_F16(Ar, Bi, ri[ct]);
            ir[ct] = WMMA_F16(Ai, Br, ir[ct]);
        }
        __syncthreads();                          // p readers done; p^1 writes visible
    }

    const int rb = rowBase + wy * 16 + hi * 8;
#pragma unroll
    for (int ct = 0; ct < 4; ++ct) {
        int col = colBase + wx * 64 + ct * 16 + m;
#pragma unroll
        for (int r = 0; r < 8; ++r) {
            size_t o = (size_t)(rb + r) * FF + col;
            Yr[o] = rr[ct][r] - ii[ct][r];        // real = rr - ii
            Yi[o] = ri[ct][r] + ir[ct][r];        // imag = ri + ir
        }
    }
}

// Z2 = 2*(Lc@Z1) - Z0, in: U (f32, from k_cgemm_lc), out: f16 row-major
__global__ void k_z2(const float* __restrict__ Z0r, const float* __restrict__ Z0i,
                     const float* __restrict__ Ur,  const float* __restrict__ Ui,
                     _Float16* __restrict__ rmR, _Float16* __restrict__ rmI) {
    int idx = blockIdx.x * 256 + threadIdx.x;
    rmR[idx] = (_Float16)(2.f * Ur[idx] - Z0r[idx]);
    rmI[idx] = (_Float16)(2.f * Ui[idx] - Z0i[idx]);
}

// W[t][k][n] (f32) -> Wt[t][n][k] (f16, column-major for the B operand)
__global__ void k_prepw(const float* __restrict__ W, _Float16* __restrict__ Wt) {
    int idx = blockIdx.x * 256 + threadIdx.x;         // 3*256*256
    int t = idx >> 16, rem = idx & 65535;
    int k = rem >> 8, n = rem & 255;
    Wt[t * 65536 + n * 256 + k] = (_Float16)W[idx];
}

// S = Z0@W0 + Z1@W1 + Z2@W2 (W real). Yc = i*S => outR = -Si + b, outI = Sr + b
__global__ void k_cgemm_zw(const _Float16* __restrict__ Z0r, const _Float16* __restrict__ Z0i,
                           const _Float16* __restrict__ Z1r, const _Float16* __restrict__ Z1i,
                           const _Float16* __restrict__ Z2r, const _Float16* __restrict__ Z2i,
                           const _Float16* __restrict__ Wt,  const float* __restrict__ bias,
                           float* __restrict__ Or, float* __restrict__ Oi) {
    int lane = threadIdx.x & 31;
    int wave = threadIdx.x >> 5;
    int tile = blockIdx.x * 4 + wave;
    int tCol = tile & 15;
    int tRow = tile >> 4;
    int m    = lane & 15;
    int hi   = lane >> 4;
    int aoff = hi * 8, boff = hi * 8;

    const _Float16* Zr[3] = { Z0r, Z1r, Z2r };
    const _Float16* Zi[3] = { Z0i, Z1i, Z2i };
    size_t arow = (size_t)(tRow * 16 + m) * 256;
    size_t brow = (size_t)(tCol * 16 + m) * 256;

    v8f sr = {}, si = {};
#pragma unroll
    for (int t = 0; t < 3; ++t) {
        const _Float16* ar = Zr[t] + arow;
        const _Float16* ai = Zi[t] + arow;
        const _Float16* bw = Wt + t * 65536 + brow;
        for (int k0 = 0; k0 < 256; k0 += 32) {
            v16h Ar = frag2(ar + k0 + aoff, ar + k0 + aoff + 16);
            v16h Ai = frag2(ai + k0 + aoff, ai + k0 + aoff + 16);
            v16h Bw = frag2(bw + k0 + boff, bw + k0 + boff + 16);
            sr = WMMA_F16(Ar, Bw, sr);
            si = WMMA_F16(Ai, Bw, si);
        }
    }
    int col = tCol * 16 + m;
    int rb  = tRow * 16 + hi * 8;
    float b = bias[col];
#pragma unroll
    for (int r = 0; r < 8; ++r) {
        size_t o = (size_t)(rb + r) * FF + col;
        Or[o] = -si[r] + b;       // Yc.real = -Si
        Oi[o] =  sr[r] + b;       // Yc.imag =  Sr
    }
}

// ---------------------------------------------------------------------------
// Classifier: logits = [r|i] @ Wc^T + bc, then log_softmax. One block per row.
// ---------------------------------------------------------------------------

__global__ void k_cls(const float* __restrict__ Xr, const float* __restrict__ Xi,
                      const float* __restrict__ Wc, const float* __restrict__ bc,
                      float* __restrict__ out) {
    __shared__ float xs[512];
    __shared__ float lg[64];
    __shared__ float red[2];
    int row = blockIdx.x;
    int tid = threadIdx.x;                             // 64 threads
    for (int j = tid; j < 256; j += 64) {
        xs[j]       = Xr[(size_t)row * 256 + j];
        xs[256 + j] = Xi[(size_t)row * 256 + j];
    }
    __syncthreads();
    if (tid < CC) {
        const float* w = Wc + (size_t)tid * 512;
        float acc = bc[tid];
        for (int j = 0; j < 512; ++j) acc += xs[j] * w[j];
        lg[tid] = acc;
    }
    __syncthreads();
    if (tid == 0) {
        float mx = lg[0];
        for (int c = 1; c < CC; ++c) mx = fmaxf(mx, lg[c]);
        float s = 0.f;
        for (int c = 0; c < CC; ++c) s += expf(lg[c] - mx);
        red[0] = mx;
        red[1] = logf(s);
    }
    __syncthreads();
    if (tid < CC) out[(size_t)row * CC + tid] = lg[tid] - red[0] - red[1];
}

// ---------------------------------------------------------------------------
// Host launcher
// ---------------------------------------------------------------------------

extern "C" void kernel_launch(void* const* d_in, const int* in_sizes, int n_in,
                              void* d_out, int out_size, void* d_ws, size_t ws_size,
                              hipStream_t stream) {
    const float* real = (const float*)d_in[0];
    const float* imag = (const float*)d_in[1];
    const int*   edges = (const int*)d_in[2];
    const float* qp   = (const float*)d_in[3];
    const float* ew   = (const float*)d_in[4];
    const float* W1   = (const float*)d_in[5];
    const float* b1   = (const float*)d_in[6];
    const float* W2   = (const float*)d_in[7];
    const float* b2   = (const float*)d_in[8];
    const float* Wc   = (const float*)d_in[9];
    const float* bc   = (const float*)d_in[10];
    float* out = (float*)d_out;
    const int E = in_sizes[2] / 2;

    char* ws = (char*)d_ws;
    const size_t MB = 1u << 20;
    _Float16* LcR = (_Float16*)(ws);                 // 32 MiB
    _Float16* LcI = (_Float16*)(ws + 32 * MB);       // 32 MiB
    char* R = ws + 64 * MB;                          // 64 MiB region: A first, then Z's
    float* A = (float*)R;                            // N*N f32 (dead after k_build_lc)
    float*    Z1r32 = (float*)(R + 0 * MB);
    float*    Z1i32 = (float*)(R + 4 * MB);
    float*    Ur32  = (float*)(R + 8 * MB);          // raw Lc@Z1 before Z2 combine
    float*    Ui32  = (float*)(R + 12 * MB);
    _Float16* Z0rmR = (_Float16*)(R + 16 * MB);
    _Float16* Z0rmI = (_Float16*)(R + 18 * MB);
    _Float16* Z0cmR = (_Float16*)(R + 20 * MB);
    _Float16* Z0cmI = (_Float16*)(R + 22 * MB);
    _Float16* Z1rmR = (_Float16*)(R + 24 * MB);
    _Float16* Z1rmI = (_Float16*)(R + 26 * MB);
    _Float16* Z1cmR = (_Float16*)(R + 28 * MB);
    _Float16* Z1cmI = (_Float16*)(R + 30 * MB);
    _Float16* Z2rmR = (_Float16*)(R + 32 * MB);
    _Float16* Z2rmI = (_Float16*)(R + 34 * MB);
    float*    O1r   = (float*)(R + 36 * MB);
    float*    O1i   = (float*)(R + 40 * MB);
    float*    O2r   = (float*)(R + 44 * MB);
    float*    O2i   = (float*)(R + 48 * MB);
    _Float16* Wt16  = (_Float16*)(R + 52 * MB);      // 384 KiB
    float* rowsum = (float*)(ws + 128 * MB);
    float* dinv   = rowsum + NN;

    const int NF_BLK = (NN * FF) / 256;              // 4096
    const int LC_BLK = (NN / 64) * 2;                // 128 blocks (64x128 tiles)
    const int ZW_BLK = (NN / 16) * (FF / 16) / 4;    // 1024 blocks, 4 waves each

    // ---- Lc construction ----
    hipMemsetAsync(A, 0, (size_t)NN * NN * sizeof(float), stream);
    k_scatter<<<(E + 255) / 256, 256, 0, stream>>>(edges, ew, A, E);
    k_rowsum<<<NN, 256, 0, stream>>>(A, rowsum);
    k_dinv<<<NN / 256, 256, 0, stream>>>(A, rowsum, dinv);
    k_build_lc<<<(NN / 256) * NN, 256, 0, stream>>>(A, dinv, qp, LcR, LcI);

    // ---- layer 1 ----
    k_cvt<<<NF_BLK, 256, 0, stream>>>(real, imag, Z0rmR, Z0rmI, Z0cmR, Z0cmI);
    k_cgemm_lc<<<LC_BLK, 256, 0, stream>>>(LcR, LcI, Z0cmR, Z0cmI, Z1r32, Z1i32);
    k_cvt<<<NF_BLK, 256, 0, stream>>>(Z1r32, Z1i32, Z1rmR, Z1rmI, Z1cmR, Z1cmI);
    k_cgemm_lc<<<LC_BLK, 256, 0, stream>>>(LcR, LcI, Z1cmR, Z1cmI, Ur32, Ui32);
    k_z2<<<NF_BLK, 256, 0, stream>>>(real, imag, Ur32, Ui32, Z2rmR, Z2rmI);
    k_prepw<<<3 * 256, 256, 0, stream>>>(W1, Wt16);
    k_cgemm_zw<<<ZW_BLK, 128, 0, stream>>>(Z0rmR, Z0rmI, Z1rmR, Z1rmI, Z2rmR, Z2rmI,
                                           Wt16, b1, O1r, O1i);

    // ---- layer 2 ----
    k_cvt<<<NF_BLK, 256, 0, stream>>>(O1r, O1i, Z0rmR, Z0rmI, Z0cmR, Z0cmI);
    k_cgemm_lc<<<LC_BLK, 256, 0, stream>>>(LcR, LcI, Z0cmR, Z0cmI, Z1r32, Z1i32);
    k_cvt<<<NF_BLK, 256, 0, stream>>>(Z1r32, Z1i32, Z1rmR, Z1rmI, Z1cmR, Z1cmI);
    k_cgemm_lc<<<LC_BLK, 256, 0, stream>>>(LcR, LcI, Z1cmR, Z1cmI, Ur32, Ui32);
    k_z2<<<NF_BLK, 256, 0, stream>>>(O1r, O1i, Ur32, Ui32, Z2rmR, Z2rmI);
    k_prepw<<<3 * 256, 256, 0, stream>>>(W2, Wt16);
    k_cgemm_zw<<<ZW_BLK, 128, 0, stream>>>(Z0rmR, Z0rmI, Z1rmR, Z1rmI, Z2rmR, Z2rmI,
                                           Wt16, b2, O2r, O2i);

    // ---- classifier + log_softmax ----
    k_cls<<<NN, 64, 0, stream>>>(O2r, O2i, Wc, bc, out);
}
